// HighResReEncoder_2688649527333
// MI455X (gfx1250) — matrix operand order repaired and checked by reading, not compiled, not run on hardware
//
#include <hip/hip_runtime.h>
#include <hip/hip_bf16.h>
#include <math.h>
#include <stdint.h>

// ---------------------------------------------------------------------------
// HighResReEncoder for MI455X (gfx1250): bf16 WMMA GEMM pipeline, f32 accum.
// Round 2: pre-transposed bf16 weights, v_cvt_pk_bf16_f32, LDS double-buffer,
//          async global->LDS loads (ASYNCcnt) for the bf16 A operand.
// ---------------------------------------------------------------------------

typedef __attribute__((ext_vector_type(16))) __bf16 v16bf;
typedef __attribute__((ext_vector_type(2)))  __bf16 v2bf;
typedef __attribute__((ext_vector_type(8)))  float  v8f;
typedef unsigned short u16;
typedef unsigned int   u32;

#define TOKEN_DIM 768
#define HIDDEN    384
#define BATCH     32
#define TSEQ      1024
#define MTOT      (BATCH * TSEQ)   // 32768
#define TOPK      154              // round(1024 * 0.15)
#define LDS_STRIDE 40              // 32 data halves + 8 pad (80B rows, 16B aligned)

// ---- bf16 helpers ---------------------------------------------------------
__device__ __forceinline__ u16 f2bf(float x) {
  union { float f; u32 u; } v; v.f = x;
  u32 r = v.u + 0x7FFFu + ((v.u >> 16) & 1u);
  return (u16)(r >> 16);
}
__device__ __forceinline__ float bf2f(u16 h) {
  union { u32 u; float f; } v; v.u = ((u32)h) << 16;
  return v.f;
}
// pack two f32 -> packed bf16x2 (single v_cvt_pk_bf16_f32 when available)
__device__ __forceinline__ u32 pack2bf(float x, float y) {
#if defined(__has_builtin) && __has_builtin(__builtin_amdgcn_cvt_pk_bf16_f32)
  union { v2bf v; u32 u; } c;
  c.v = __builtin_amdgcn_cvt_pk_bf16_f32(x, y);
  return c.u;
#else
  return (u32)f2bf(x) | ((u32)f2bf(y) << 16);
#endif
}
__device__ __forceinline__ void wait_async0() {
#if defined(__has_builtin) && __has_builtin(__builtin_amdgcn_s_wait_asynccnt)
  __builtin_amdgcn_s_wait_asynccnt(0);
#else
  asm volatile("s_wait_asynccnt 0x0" ::: "memory");
#endif
}

// ---- WMMA fragment loads from LDS -----------------------------------------
// A (16x32 bf16, MxK): lane L holds row m=L&15; half j -> K = j + 8*(j>=8) + 8*hi
__device__ __forceinline__ v16bf load_a_frag(const u16* rowBase, int hi) {
  union { uint4 u[2]; v16bf v; } f;
  f.u[0] = *(const uint4*)(rowBase + 8 * hi);        // K = 8*hi .. 8*hi+7
  f.u[1] = *(const uint4*)(rowBase + 16 + 8 * hi);   // K = 16+8*hi .. 23+8*hi
  return f.v;
}
// B (32x16 bf16, KxN) stored transposed in LDS as [n][k]:
// lane L holds col n=L&15; half j -> K = j + 8*(j>=8) + 8*hi
__device__ __forceinline__ v16bf load_b_frag(const u16* rowBase, int hi) {
  union { uint4 u[2]; v16bf v; } f;
  const u16* p = rowBase + 8 * hi;
  f.u[0] = *(const uint4*)(p);
  f.u[1] = *(const uint4*)(p + 16);
  return f.v;
}

// ---- one K=32 step of the 128x128 block tile (8 waves, wave tile 32x64) ---
__device__ __forceinline__ void mma_step(const u16* ldsA, const u16* ldsB,
                                         int wm, int wn, int lane, v8f acc[2][4]) {
  const int lrow = lane & 15;
  const int hi   = lane >> 4;
  v16bf a0 = load_a_frag(ldsA + (wm * 32 +  0 + lrow) * LDS_STRIDE, hi);
  v16bf a1 = load_a_frag(ldsA + (wm * 32 + 16 + lrow) * LDS_STRIDE, hi);
  v16bf b[4];
#pragma unroll
  for (int tn = 0; tn < 4; ++tn)
    b[tn] = load_b_frag(ldsB + (wn * 64 + tn * 16 + lrow) * LDS_STRIDE, hi);
#pragma unroll
  for (int tn = 0; tn < 4; ++tn) {
    acc[0][tn] = __builtin_amdgcn_wmma_f32_16x16x32_bf16(
        false, a0, false, b[tn], (short)0, acc[0][tn], false, false);
    acc[1][tn] = __builtin_amdgcn_wmma_f32_16x16x32_bf16(
        false, a1, false, b[tn], (short)0, acc[1][tn], false, false);
  }
}

__device__ __forceinline__ void zero_acc(v8f acc[2][4]) {
#pragma unroll
  for (int i = 0; i < 2; ++i)
#pragma unroll
    for (int j = 0; j < 4; ++j)
#pragma unroll
      for (int r = 0; r < 8; ++r) acc[i][j][r] = 0.0f;
}

// ---- B tile: 128 n-rows x 32 k from pre-transposed bf16 Wt[N][K] ----------
__device__ __forceinline__ void load_b_tile(u16* ldsB, const u16* __restrict__ Wt,
                                            int Ktot, int k0, int n0, int tid) {
#pragma unroll
  for (int i = 0; i < 2; ++i) {
    int lin = tid + i * 256;      // 0..511
    int r   = lin >> 2;           // n row 0..127
    int q   = (lin & 3) << 3;     // half offset 0,8,16,24 (16B aligned)
    uint4 v = *(const uint4*)(Wt + (long)(n0 + r) * Ktot + k0 + q);
    *(uint4*)&ldsB[r * LDS_STRIDE + q] = v;
  }
}

// ===========================================================================
// Kernel P: weight prep — convert f32 [K][N] to bf16 transposed [N][K]
// ===========================================================================
__global__ __launch_bounds__(256)
void wtrans_kernel(const float* __restrict__ W, u16* __restrict__ Wt,
                   int K, int N) {
  int idx = blockIdx.x * 256 + threadIdx.x;
  if (idx >= K * N) return;
  int k = idx / N, n = idx - k * N;        // coalesced read along n
  Wt[(long)n * K + k] = f2bf(W[idx]);
}

// ===========================================================================
// Kernel 0: top-k mask via rank counting (matches lax.top_k tie-breaking)
// ===========================================================================
__global__ __launch_bounds__(1024)
void mask_kernel(const float* __restrict__ scores, float* __restrict__ mask) {
  __shared__ float s[TSEQ];
  const int b = blockIdx.x, t = threadIdx.x;
  const float x = scores[b * TSEQ + t];
  s[t] = x;
  __syncthreads();
  int cnt = 0;
  for (int j = 0; j < TSEQ; ++j) {
    float y = s[j];
    cnt += (y > x) || (y == x && j < t);
  }
  mask[b * TSEQ + t] = (cnt < TOPK) ? 1.0f : 0.0f;
}

// ===========================================================================
// Kernel 1: h = GELU(local @ W1 + b1)   M=32768 N=384 K=3072 (gathered A)
// ===========================================================================
__global__ __launch_bounds__(256)
void gemm1_kernel(const float* __restrict__ highres, const u16* __restrict__ W1t,
                  const float* __restrict__ b1, u16* __restrict__ h) {
  __shared__ u16 ldsA[2][128 * LDS_STRIDE];
  __shared__ u16 ldsB[2][128 * LDS_STRIDE];
  const int tid = threadIdx.x, lane = tid & 31, wave = tid >> 5;
  const int wm = wave & 3, wn = wave >> 2;
  const int n0 = blockIdx.x * 128;   // 3 N blocks (fast dim -> L2 reuse of A)
  const int m0 = blockIdx.y * 128;   // 256 M blocks
  v8f acc[2][4];
  zero_acc(acc);

  auto load_tile = [&](int kk, int buf) {
    const int k0 = kk * 32;
    const int p  = k0 / TOKEN_DIM, off = k0 % TOKEN_DIM;
    const int sy = p >> 1, sx = p & 1;
#pragma unroll
    for (int i = 0; i < 4; ++i) {
      int lin = tid + i * 256;
      int r   = lin >> 3;
      int c   = (lin & 7) << 2;
      int m   = m0 + r;
      int bb  = m >> 10, t = m & 1023;
      int gy  = t >> 5, gx = t & 31;
      long hr = ((long)(bb * 4096 + gy * 128 + sy * 64 + gx * 2 + sx)) * TOKEN_DIM
                + off + c;
      float4 v = *(const float4*)(highres + hr);
      *(uint2*)&ldsA[buf][r * LDS_STRIDE + c] =
          make_uint2(pack2bf(v.x, v.y), pack2bf(v.z, v.w));
    }
    load_b_tile(ldsB[buf], W1t, 3072, k0, n0, tid);
  };

  load_tile(0, 0);
  __syncthreads();
  for (int kk = 0; kk < 96; ++kk) {
    const int cur = kk & 1;
    if (kk + 1 < 96) load_tile(kk + 1, cur ^ 1);
    mma_step(ldsA[cur], ldsB[cur], wm, wn, lane, acc);
    __syncthreads();
  }

  const int lrow = lane & 15, hi = lane >> 4;
#pragma unroll
  for (int tm = 0; tm < 2; ++tm)
#pragma unroll
    for (int tn = 0; tn < 4; ++tn) {
      int n = n0 + wn * 64 + tn * 16 + lrow;
      float bias = b1[n];
#pragma unroll
      for (int r = 0; r < 8; ++r) {
        int m = m0 + wm * 32 + tm * 16 + r + 8 * hi;
        float x = acc[tm][tn][r] + bias;
        float g = 0.5f * x * (1.0f + erff(x * 0.70710678118654752f));
        h[(long)m * HIDDEN + n] = f2bf(g);
      }
    }
}

// ===========================================================================
// Kernel 2: refined_local = h @ W2 + b2   M=32768 N=768 K=384
//   A operand is already bf16 -> async global->LDS copy (ASYNCcnt path)
// ===========================================================================
__global__ __launch_bounds__(256)
void gemm2_kernel(const u16* __restrict__ h, const u16* __restrict__ W2t,
                  const float* __restrict__ b2, float* __restrict__ rl) {
  __shared__ u16 ldsA[2][128 * LDS_STRIDE];
  __shared__ u16 ldsB[2][128 * LDS_STRIDE];
  const int tid = threadIdx.x, lane = tid & 31, wave = tid >> 5;
  const int wm = wave & 3, wn = wave >> 2;
  const int n0 = blockIdx.x * 128;   // 6 N blocks
  const int m0 = blockIdx.y * 128;
  v8f acc[2][4];
  zero_acc(acc);

  auto load_tile = [&](int kk, int buf) {
    const int k0 = kk * 32;
#pragma unroll
    for (int i = 0; i < 2; ++i) {
      int lin = tid + i * 256;      // 0..511
      int r   = lin >> 2;
      int q   = (lin & 3) << 3;     // half offset 0,8,16,24 (16B aligned)
      u32 ldsoff = (u32)(uintptr_t)&ldsA[buf][r * LDS_STRIDE + q];
      const u16* g = h + (long)(m0 + r) * HIDDEN + k0 + q;
      // CDNA5 async copy: no VGPR round-trip, tracked on ASYNCcnt
      asm volatile("global_load_async_to_lds_b128 %0, %1, off"
                   :: "v"(ldsoff), "v"(g) : "memory");
    }
    load_b_tile(ldsB[buf], W2t, HIDDEN, k0, n0, tid);
  };

  load_tile(0, 0);
  wait_async0();
  __syncthreads();
  for (int kk = 0; kk < 12; ++kk) {
    const int cur = kk & 1;
    if (kk + 1 < 12) load_tile(kk + 1, cur ^ 1);
    mma_step(ldsA[cur], ldsB[cur], wm, wn, lane, acc);
    wait_async0();
    __syncthreads();
  }

  const int lrow = lane & 15, hi = lane >> 4;
#pragma unroll
  for (int tm = 0; tm < 2; ++tm)
#pragma unroll
    for (int tn = 0; tn < 4; ++tn) {
      int n = n0 + wn * 64 + tn * 16 + lrow;
      float bias = b2[n];
#pragma unroll
      for (int r = 0; r < 8; ++r) {
        int m = m0 + wm * 32 + tm * 16 + r + 8 * hi;
        rl[(long)m * TOKEN_DIM + n] = acc[tm][tn][r] + bias;
      }
    }
}

// ===========================================================================
// Kernel 3: hg = SiLU(base@Wg1[0:768] + rl@Wg1[768:1536] + score*Wg1[1536] + bg1)
//           M=32768 N=384 K=1536
// ===========================================================================
__global__ __launch_bounds__(256)
void gemm3_kernel(const float* __restrict__ base, const float* __restrict__ rl,
                  const float* __restrict__ scores, const u16* __restrict__ Wg1t,
                  const float* __restrict__ Wg1, const float* __restrict__ bg1,
                  u16* __restrict__ hg) {
  __shared__ u16 ldsA[2][128 * LDS_STRIDE];
  __shared__ u16 ldsB[2][128 * LDS_STRIDE];
  const int tid = threadIdx.x, lane = tid & 31, wave = tid >> 5;
  const int wm = wave & 3, wn = wave >> 2;
  const int n0 = blockIdx.x * 128;   // 3 N blocks
  const int m0 = blockIdx.y * 128;
  v8f acc[2][4];
  zero_acc(acc);

  auto load_tile = [&](int kk, int buf) {
    const int k0 = kk * 32;
    const float* src = (k0 < TOKEN_DIM) ? base : rl;
    const int koff = (k0 < TOKEN_DIM) ? k0 : (k0 - TOKEN_DIM);
#pragma unroll
    for (int i = 0; i < 4; ++i) {
      int lin = tid + i * 256;
      int r   = lin >> 3;
      int c   = (lin & 7) << 2;
      float4 v = *(const float4*)(src + (long)(m0 + r) * TOKEN_DIM + koff + c);
      *(uint2*)&ldsA[buf][r * LDS_STRIDE + c] =
          make_uint2(pack2bf(v.x, v.y), pack2bf(v.z, v.w));
    }
    load_b_tile(ldsB[buf], Wg1t, 1536, k0, n0, tid);
  };

  load_tile(0, 0);
  __syncthreads();
  for (int kk = 0; kk < 48; ++kk) {
    const int cur = kk & 1;
    if (kk + 1 < 48) load_tile(kk + 1, cur ^ 1);
    mma_step(ldsA[cur], ldsB[cur], wm, wn, lane, acc);
    __syncthreads();
  }

  const int lrow = lane & 15, hi = lane >> 4;
  const float* wScore = Wg1 + (long)1536 * HIDDEN;  // score-feature row (f32)
#pragma unroll
  for (int tm = 0; tm < 2; ++tm)
#pragma unroll
    for (int tn = 0; tn < 4; ++tn) {
      int n = n0 + wn * 64 + tn * 16 + lrow;
      float bias = bg1[n];
      float ws   = wScore[n];
#pragma unroll
      for (int r = 0; r < 8; ++r) {
        int m = m0 + wm * 32 + tm * 16 + r + 8 * hi;
        float x = acc[tm][tn][r] + bias + scores[m] * ws;
        float s = x / (1.0f + expf(-x));   // SiLU
        hg[(long)m * HIDDEN + n] = f2bf(s);
      }
    }
}

// ===========================================================================
// Kernel 4: mg = mask * sigmoid(hg @ Wg2 + bg2)   (wave-per-row mat-vec)
// ===========================================================================
__global__ __launch_bounds__(256)
void gate_kernel(const u16* __restrict__ hg, const float* __restrict__ Wg2,
                 const float* __restrict__ bg2, const float* __restrict__ mask,
                 float* __restrict__ mg) {
  const int row  = blockIdx.x * 8 + (threadIdx.x >> 5);
  const int lane = threadIdx.x & 31;
  const u16* hr = hg + (long)row * HIDDEN;
  float sum = 0.0f;
#pragma unroll
  for (int j = lane; j < HIDDEN; j += 32)
    sum += bf2f(hr[j]) * Wg2[j];
#pragma unroll
  for (int o = 16; o > 0; o >>= 1) sum += __shfl_down(sum, o, 32);
  if (lane == 0) {
    float gate = 1.0f / (1.0f + expf(-(sum + bg2[0])));
    mg[row] = gate * mask[row];
  }
}

// ===========================================================================
// Kernel 5: out = base + mg * (refined_local - base)   (float4)
// ===========================================================================
__global__ __launch_bounds__(256)
void blend_kernel(const float* __restrict__ base, const float* __restrict__ rl,
                  const float* __restrict__ mg, float* __restrict__ out) {
  const int i = blockIdx.x * 256 + threadIdx.x;     // float4 index
  const int row = i / (TOKEN_DIM / 4);              // 192 float4 per row
  const float m = mg[row];
  float4 b = ((const float4*)base)[i];
  float4 r = ((const float4*)rl)[i];
  float4 o;
  o.x = b.x + m * (r.x - b.x);
  o.y = b.y + m * (r.y - b.y);
  o.z = b.z + m * (r.z - b.z);
  o.w = b.w + m * (r.w - b.w);
  ((float4*)out)[i] = o;
}

// ===========================================================================
extern "C" void kernel_launch(void* const* d_in, const int* in_sizes, int n_in,
                              void* d_out, int out_size, void* d_ws, size_t ws_size,
                              hipStream_t stream) {
  const float* base    = (const float*)d_in[0];
  const float* highres = (const float*)d_in[1];
  const float* scores  = (const float*)d_in[2];
  const float* W1      = (const float*)d_in[3];
  const float* b1      = (const float*)d_in[4];
  const float* W2      = (const float*)d_in[5];
  const float* b2      = (const float*)d_in[6];
  const float* Wg1     = (const float*)d_in[7];
  const float* bg1     = (const float*)d_in[8];
  const float* Wg2     = (const float*)d_in[9];
  const float* bg2     = (const float*)d_in[10];
  float* out = (float*)d_out;

  char* ws = (char*)d_ws;
  size_t o = 0;
  const size_t sz_h    = (size_t)MTOT * HIDDEN * sizeof(u16);      // 24 MB
  const size_t sz_rl   = (size_t)MTOT * TOKEN_DIM * sizeof(float); // 96 MB
  const size_t sz_hg   = (size_t)MTOT * HIDDEN * sizeof(u16);      // 24 MB
  const size_t sz_msk  = (size_t)MTOT * sizeof(float);
  const size_t sz_w1t  = (size_t)3072 * 384 * sizeof(u16);
  const size_t sz_w2t  = (size_t)384 * 768 * sizeof(u16);
  const size_t sz_wg1t = (size_t)1536 * 384 * sizeof(u16);
  u16*   h    = (u16*)(ws + o);   o += sz_h;
  float* rl   = (float*)(ws + o); o += sz_rl;
  u16*   hg   = (u16*)(ws + o);   o += sz_hg;
  float* mask = (float*)(ws + o); o += sz_msk;
  float* mg   = (float*)(ws + o); o += sz_msk;
  u16*   W1t  = (u16*)(ws + o);   o += sz_w1t;
  u16*   W2t  = (u16*)(ws + o);   o += sz_w2t;
  u16*   Wg1t = (u16*)(ws + o);   o += sz_wg1t;

  // weight prep (bf16 + transpose); score row of Wg1 stays f32 for epilogue
  wtrans_kernel<<<(3072 * 384 + 255) / 256, 256, 0, stream>>>(W1, W1t, 3072, 384);
  wtrans_kernel<<<(384 * 768 + 255) / 256, 256, 0, stream>>>(W2, W2t, 384, 768);
  wtrans_kernel<<<(1536 * 384 + 255) / 256, 256, 0, stream>>>(Wg1, Wg1t, 1536, 384);

  mask_kernel<<<BATCH, TSEQ, 0, stream>>>(scores, mask);
  gemm1_kernel<<<dim3(3, MTOT / 128), 256, 0, stream>>>(highres, W1t, b1, h);
  gemm2_kernel<<<dim3(6, MTOT / 128), 256, 0, stream>>>(h, W2t, b2, rl);
  gemm3_kernel<<<dim3(3, MTOT / 128), 256, 0, stream>>>(base, rl, scores, Wg1t, Wg1, bg1, hg);
  gate_kernel<<<MTOT / 8, 256, 0, stream>>>(hg, Wg2, bg2, mask, mg);
  blend_kernel<<<(MTOT * TOKEN_DIM / 4) / 256, 256, 0, stream>>>(base, rl, mg, out);
}
